// GRU_model_56487409877294
// MI455X (gfx1250) — matrix operand order repaired and checked by reading, not compiled, not run on hardware
//
#include <hip/hip_runtime.h>
#include <hip/hip_bf16.h>
#include <math.h>

#define DIN   1024
#define HH    512
#define BB    32768

#define KSTEP 32
#define MTILE 128
#define NTILE 128
#define LDSTR 40          // padded LDS K-stride (80B rows)
#define WT_STRIDE 1536    // per-column K extent of fused [W;U] transposed weights
#define WT_GATE   (512 * WT_STRIDE)

typedef __attribute__((ext_vector_type(16))) __bf16 v16bf;
typedef __attribute__((ext_vector_type(8)))  __bf16 v8bf;
typedef __attribute__((ext_vector_type(4)))  __bf16 v4bf;
typedef __attribute__((ext_vector_type(8)))  float  v8f;

__device__ __forceinline__ v16bf frag_cat(const __bf16* p0, const __bf16* p1) {
    v8bf lo = *(const v8bf*)p0;
    v8bf hi = *(const v8bf*)p1;
    return __builtin_shufflevector(lo, hi, 0,1,2,3,4,5,6,7,8,9,10,11,12,13,14,15);
}

__device__ __forceinline__ float sigmoidf_(float v) {
    return 1.0f / (1.0f + __expf(-v));
}

// CDNA5 async global->LDS copy (ASYNCcnt-tracked, no VGPR round trip).
// vdst = LDS byte offset (low 32 bits of generic shared pointer), vaddr = global.
__device__ __forceinline__ void async_copy_b128(void* lds_dst, const void* gsrc) {
    unsigned           l = (unsigned)(uintptr_t)lds_dst;
    unsigned long long g = (unsigned long long)(uintptr_t)gsrc;
    asm volatile("global_load_async_to_lds_b128 %0, %1, off"
                 :: "v"(l), "v"(g) : "memory");
}
__device__ __forceinline__ void wait_asynccnt0() {
    asm volatile("s_wait_asynccnt 0x0" ::: "memory");
}

// ---------------------------------------------------------------------------
// Setup A: elementwise fp32 -> bf16 (b128 in, b128 out; 8 elems/thread)
// ---------------------------------------------------------------------------
__global__ __launch_bounds__(256)
void cvt_f32_bf16(const float* __restrict__ src, __bf16* __restrict__ dst) {
    size_t i = ((size_t)blockIdx.x * 256 + threadIdx.x) * 8;
    float4 a = *(const float4*)(src + i);
    float4 b = *(const float4*)(src + i + 4);
    v8bf o;
    o[0] = (__bf16)a.x; o[1] = (__bf16)a.y; o[2] = (__bf16)a.z; o[3] = (__bf16)a.w;
    o[4] = (__bf16)b.x; o[5] = (__bf16)b.y; o[6] = (__bf16)b.z; o[7] = (__bf16)b.w;
    *(v8bf*)(dst + i) = o;
}

// ---------------------------------------------------------------------------
// Setup B: weight transpose-convert. src: fp32 [K x 512] row-major.
// dst: bf16 column-major, dst[n*dstStride + k]. 32x32 tile through LDS.
// ---------------------------------------------------------------------------
__global__ __launch_bounds__(256)
void wtrans_cvt(const float* __restrict__ src, __bf16* __restrict__ dst,
                int dstStride) {
    __shared__ __bf16 t[32][36];
    const int tid = threadIdx.x;
    const int nb = blockIdx.x * 32;
    const int kb = blockIdx.y * 32;

    int kk = tid >> 3, n4 = (tid & 7) * 4;
    float4 f = *(const float4*)(src + (size_t)(kb + kk) * HH + nb + n4);
    t[n4 + 0][kk] = (__bf16)f.x;
    t[n4 + 1][kk] = (__bf16)f.y;
    t[n4 + 2][kk] = (__bf16)f.z;
    t[n4 + 3][kk] = (__bf16)f.w;
    __syncthreads();

    int n = tid >> 3, k4 = (tid & 7) * 4;
    v4bf o = *(const v4bf*)&t[n][k4];
    *(v4bf*)(dst + (size_t)(nb + n) * dstStride + kb + k4) = o;
}

// ---------------------------------------------------------------------------
// GEMM core pieces
// ---------------------------------------------------------------------------
__device__ __forceinline__ void stage_async(
    const __bf16* __restrict__ Arow0, int Astride, int kA,
    const __bf16* __restrict__ Bcol0, int Bstride, int kB,
    __bf16* bufA, __bf16* bufB, int tid)
{
    // A tile: 128 rows x 32 k (8KB) ; 2 x b128 per thread
    #pragma unroll
    for (int c = 0; c < 2; ++c) {
        int e = c * 256 + tid;
        int r = e >> 2, q = e & 3;
        async_copy_b128(&bufA[r * LDSTR + q * 8],
                        &Arow0[(size_t)r * Astride + kA + q * 8]);
    }
    // B tile: 128 cols x 32 k (8KB, already column-major) ; 2 x b128 per thread
    #pragma unroll
    for (int c = 0; c < 2; ++c) {
        int e = c * 256 + tid;
        int n = e >> 2, q = e & 3;
        async_copy_b128(&bufB[n * LDSTR + q * 8],
                        &Bcol0[(size_t)n * Bstride + kB + q * 8]);
    }
}

__device__ __forceinline__ void compute_tile(
    const __bf16* ldsA, const __bf16* ldsB, v8f (&acc)[8],
    int wave, int l16, int half)
{
    // A fragment: lane l16 = row; half selects K pairs {0..7,16..23} / {8..15,24..31}
    const __bf16* arow = &ldsA[(wave * 16 + l16) * LDSTR];
    v16bf af = frag_cat(arow + half * 8, arow + 16 + half * 8);

    // Preload all 8 B fragments as one LDS clause, then drain DScnt across
    // the WMMA sequence instead of a full s_wait_dscnt 0 per WMMA.
    v16bf bf[8];
    #pragma unroll
    for (int j = 0; j < 8; ++j) {
        const __bf16* bcol = &ldsB[(j * 16 + l16) * LDSTR];
        bf[j] = frag_cat(bcol + half * 8, bcol + 16 + half * 8);
    }
    #pragma unroll
    for (int j = 0; j < 8; ++j) {
        acc[j] = __builtin_amdgcn_wmma_f32_16x16x32_bf16(
            false, af, false, bf[j], (short)0, acc[j], false, false);
    }
}

// Double-buffered, async-staged K-segment.
__device__ __forceinline__ void gemm_segment(
    const __bf16* __restrict__ Arow0, int Astride,
    const __bf16* __restrict__ Bcol0, int Bstride, int kmax,
    __bf16* ldsA0, __bf16* ldsA1, __bf16* ldsB0, __bf16* ldsB1,
    v8f (&acc)[8], int tid, int wave, int l16, int half)
{
    __syncthreads();   // other waves may still be reading the buffer we refill
    stage_async(Arow0, Astride, 0, Bcol0, Bstride, 0, ldsA0, ldsB0, tid);
    for (int k0 = 0; k0 < kmax; k0 += KSTEP) {
        int parity = (k0 >> 5) & 1;
        __bf16* cA = parity ? ldsA1 : ldsA0;
        __bf16* cB = parity ? ldsB1 : ldsB0;
        wait_asynccnt0();      // my async writes of the current tile are done
        __syncthreads();       // everyone's are; prior reads of next buffer done
        if (k0 + KSTEP < kmax) {
            __bf16* nA = parity ? ldsA0 : ldsA1;
            __bf16* nB = parity ? ldsB0 : ldsB1;
            stage_async(Arow0, Astride, k0 + KSTEP,
                        Bcol0, Bstride, k0 + KSTEP, nA, nB, tid);
        }
        compute_tile(cA, cB, acc, wave, l16, half);
    }
}

// ---------------------------------------------------------------------------
// Phase 1: acc = [x|h] @ [Wg;Ug].
//   gate 0 -> ws_z  = sigmoid(acc + bz + buz)
//   gate 1 -> ws_rh = sigmoid(acc + br + bur) * h_old   (bf16)
//   gate 2 -> ws_xh = acc + bh
// ---------------------------------------------------------------------------
__global__ __launch_bounds__(256)
void gru_phase1(const __bf16* __restrict__ x_bf, const __bf16* __restrict__ h_bf,
                const __bf16* __restrict__ WT,
                const float* __restrict__ bz, const float* __restrict__ buz,
                const float* __restrict__ br, const float* __restrict__ bur,
                const float* __restrict__ bh,
                float* __restrict__ ws_z, __bf16* __restrict__ ws_rh,
                float* __restrict__ ws_xh) {
    __shared__ __align__(16) __bf16 ldsA0[MTILE * LDSTR];
    __shared__ __align__(16) __bf16 ldsA1[MTILE * LDSTR];
    __shared__ __align__(16) __bf16 ldsB0[NTILE * LDSTR];
    __shared__ __align__(16) __bf16 ldsB1[NTILE * LDSTR];

    const int tid  = threadIdx.x;
    const int wave = tid >> 5;
    const int lane = tid & 31;
    const int l16  = lane & 15;
    const int half = lane >> 4;

    const int mBase   = blockIdx.x * MTILE;
    const int gate    = blockIdx.y >> 2;               // 0:z 1:r 2:h-cand
    const int colBase = (blockIdx.y & 3) * NTILE;

    const v8f vzero = {0.f, 0.f, 0.f, 0.f, 0.f, 0.f, 0.f, 0.f};
    v8f acc[8];
    #pragma unroll
    for (int j = 0; j < 8; ++j) acc[j] = vzero;

    const __bf16* Ax = x_bf + (size_t)mBase * DIN;
    const __bf16* Bg = WT + (size_t)gate * WT_GATE + (size_t)colBase * WT_STRIDE;

    gemm_segment(Ax, DIN, Bg, WT_STRIDE, DIN,
                 ldsA0, ldsA1, ldsB0, ldsB1, acc, tid, wave, l16, half);

    if (gate < 2) {
        const __bf16* Ah = h_bf + (size_t)mBase * HH;
        const __bf16* Bu = Bg + DIN;                   // U part lives at k >= 1024
        gemm_segment(Ah, HH, Bu, WT_STRIDE, HH,
                     ldsA0, ldsA1, ldsB0, ldsB1, acc, tid, wave, l16, half);
    }

    // Epilogue. C layout: VGPR i -> lanes 0-15 M=i, lanes 16-31 M=i+8; N=l16.
    #pragma unroll
    for (int j = 0; j < 8; ++j) {
        int col = colBase + j * 16 + l16;
        #pragma unroll
        for (int i = 0; i < 8; ++i) {
            int row = mBase + wave * 16 + half * 8 + i;
            size_t idx = (size_t)row * HH + col;
            float v = acc[j][i];
            if (gate == 0) {
                ws_z[idx] = sigmoidf_(v + bz[col] + buz[col]);
            } else if (gate == 1) {
                float r = sigmoidf_(v + br[col] + bur[col]);
                ws_rh[idx] = (__bf16)(r * (float)h_bf[idx]);
            } else {
                ws_xh[idx] = v + bh[col];
            }
        }
    }
}

// ---------------------------------------------------------------------------
// Phase 2: acc = rh @ Uh ; h_t = z*h_old + (1-z)*tanh(xh + acc + buh)
// ---------------------------------------------------------------------------
__global__ __launch_bounds__(256)
void gru_phase2(const __bf16* __restrict__ rh, const __bf16* __restrict__ UhT,
                const float* __restrict__ buh, const float* __restrict__ ws_xh,
                const float* __restrict__ ws_z, const float* __restrict__ h_old,
                float* __restrict__ out) {
    __shared__ __align__(16) __bf16 ldsA0[MTILE * LDSTR];
    __shared__ __align__(16) __bf16 ldsA1[MTILE * LDSTR];
    __shared__ __align__(16) __bf16 ldsB0[NTILE * LDSTR];
    __shared__ __align__(16) __bf16 ldsB1[NTILE * LDSTR];

    const int tid  = threadIdx.x;
    const int wave = tid >> 5;
    const int lane = tid & 31;
    const int l16  = lane & 15;
    const int half = lane >> 4;

    const int mBase   = blockIdx.x * MTILE;
    const int colBase = blockIdx.y * NTILE;

    const v8f vzero = {0.f, 0.f, 0.f, 0.f, 0.f, 0.f, 0.f, 0.f};
    v8f acc[8];
    #pragma unroll
    for (int j = 0; j < 8; ++j) acc[j] = vzero;

    const __bf16* Ar = rh + (size_t)mBase * HH;
    const __bf16* Bu = UhT + (size_t)colBase * HH;

    gemm_segment(Ar, HH, Bu, HH, HH,
                 ldsA0, ldsA1, ldsB0, ldsB1, acc, tid, wave, l16, half);

    #pragma unroll
    for (int j = 0; j < 8; ++j) {
        int col = colBase + j * 16 + l16;
        #pragma unroll
        for (int i = 0; i < 8; ++i) {
            int row = mBase + wave * 16 + half * 8 + i;
            size_t idx = (size_t)row * HH + col;
            float v  = acc[j][i] + buh[col] + ws_xh[idx];
            float hc = tanhf(v);
            float z  = ws_z[idx];
            out[idx] = z * h_old[idx] + (1.0f - z) * hc;
        }
    }
}

// ---------------------------------------------------------------------------
extern "C" void kernel_launch(void* const* d_in, const int* in_sizes, int n_in,
                              void* d_out, int out_size, void* d_ws, size_t ws_size,
                              hipStream_t stream) {
    const float* x     = (const float*)d_in[0];
    const float* h_old = (const float*)d_in[1];
    const float* Wz    = (const float*)d_in[2];
    const float* bz    = (const float*)d_in[3];
    const float* Uz    = (const float*)d_in[4];
    const float* buz   = (const float*)d_in[5];
    const float* Wr    = (const float*)d_in[6];
    const float* br    = (const float*)d_in[7];
    const float* Ur    = (const float*)d_in[8];
    const float* bur   = (const float*)d_in[9];
    const float* Wh    = (const float*)d_in[10];
    const float* bh    = (const float*)d_in[11];
    const float* Uh    = (const float*)d_in[12];
    const float* buh   = (const float*)d_in[13];

    const size_t NBH = (size_t)BB * HH;        // 16M
    // Workspace layout (bytes): z 64M | xh 64M | rh 32M | x_bf 64M | h_bf 32M
    //                           | WT 4.5M | UhT 0.5M  => ~261 MB
    float*  ws_z  = (float*)d_ws;
    float*  ws_xh = ws_z + NBH;
    __bf16* ws_rh = (__bf16*)(ws_xh + NBH);
    __bf16* x_bf  = ws_rh + NBH;
    __bf16* h_bf  = x_bf + (size_t)BB * DIN;
    __bf16* WT    = h_bf + NBH;                // 3 gates x [512][1536]
    __bf16* UhT   = WT + (size_t)3 * WT_GATE;  // [512][512]

    dim3 blk(256);

    // --- setup: activations to bf16 ---
    cvt_f32_bf16<<<dim3(((size_t)BB * DIN) / 2048), blk, 0, stream>>>(x, x_bf);
    cvt_f32_bf16<<<dim3(NBH / 2048), blk, 0, stream>>>(h_old, h_bf);

    // --- setup: weights to bf16, transposed column-major [n][k] ---
    wtrans_cvt<<<dim3(16, 32), blk, 0, stream>>>(Wz, WT + 0 * WT_GATE,       WT_STRIDE);
    wtrans_cvt<<<dim3(16, 16), blk, 0, stream>>>(Uz, WT + 0 * WT_GATE + DIN, WT_STRIDE);
    wtrans_cvt<<<dim3(16, 32), blk, 0, stream>>>(Wr, WT + 1 * WT_GATE,       WT_STRIDE);
    wtrans_cvt<<<dim3(16, 16), blk, 0, stream>>>(Ur, WT + 1 * WT_GATE + DIN, WT_STRIDE);
    wtrans_cvt<<<dim3(16, 32), blk, 0, stream>>>(Wh, WT + 2 * WT_GATE,       WT_STRIDE);
    wtrans_cvt<<<dim3(16, 16), blk, 0, stream>>>(Uh, UhT,                    HH);

    // --- phase 1: fused gate GEMMs + activations ---
    gru_phase1<<<dim3(BB / MTILE, 3 * (HH / NTILE)), blk, 0, stream>>>(
        x_bf, h_bf, WT, bz, buz, br, bur, bh, ws_z, ws_rh, ws_xh);

    // --- phase 2: candidate GEMM + blend ---
    gru_phase2<<<dim3(BB / MTILE, HH / NTILE), blk, 0, stream>>>(
        ws_rh, UhT, buh, ws_xh, ws_z, h_old, (float*)d_out);
}
